// CustomGNN_61254823575978
// MI455X (gfx1250) — compile-verified, matrix operand order checked by reading, not run on hardware
//
#include <hip/hip_runtime.h>
#include <hip/hip_bf16.h>

typedef __attribute__((ext_vector_type(16))) __bf16 v16bf;
typedef __attribute__((ext_vector_type(8)))  float  v8f;
typedef __attribute__((ext_vector_type(4))) unsigned int tdm_u32x4;
typedef __attribute__((ext_vector_type(8))) int          tdm_i32x8;
typedef __attribute__((ext_vector_type(4))) int          tdm_i32x4;

#define D_IN   128
#define FEAT   384   // 3*D
#define H1DIM  512
#define H2DIM  64
#define KT1    12    // 384/32
#define NT1    32    // 512/16
#define KT2    16    // 512/32
#define NT2    4     // 64/16
#define KT3    2     // 64/32
#define NT3    8     // 128/16
#define WAVES  4     // waves (16-edge tiles) per block
#define NB2H   (KT2*NT2*512)   // 32768 halves = 64KB
#define NB3H   (KT3*NT3*512)   //  8192 halves = 16KB

// round-to-nearest-even f32 -> bf16
__device__ __forceinline__ unsigned short f2bf(float f) {
  unsigned int u = __float_as_uint(f);
  u += 0x7FFFu + ((u >> 16) & 1u);
  return (unsigned short)(u >> 16);
}

// B operand: 16 contiguous halves per lane (pre-swizzled) -> two b128 loads
__device__ __forceinline__ v16bf ld_b(const unsigned short* p) {
  union { uint4 q[2]; v16bf v; } u;
  u.q[0] = *(const uint4*)(p);
  u.q[1] = *(const uint4*)(p + 8);
  return u.v;
}

// A operand from LDS row: halves 0..7 = K base..base+7, halves 8..15 = K base+16..base+23
__device__ __forceinline__ v16bf ld_a(const unsigned short* p) {
  union { uint4 q[2]; v16bf v; } u;
  u.q[0] = *(const uint4*)(p);
  u.q[1] = *(const uint4*)(p + 16);
  return u.v;
}

__device__ __forceinline__ v8f wmma_bf16(v16bf a, v16bf b, v8f c) {
  return __builtin_amdgcn_wmma_f32_16x16x32_bf16(false, a, false, b, (short)0, c, false, false);
}

// ---- TDM: 1-D DMA copy of `ndwords` dwords from global to LDS (ISA 08 §8) ----
__device__ __forceinline__ void tdm_copy_1d(const unsigned short* gsrc,
                                            unsigned short* ldst,
                                            unsigned ndwords) {
  unsigned long long g = (unsigned long long)gsrc;
  unsigned lds = (unsigned)(unsigned long long)
      ( __attribute__((address_space(3))) const void*)ldst;
  tdm_u32x4 g0;
  g0[0] = 1u;                                               // count=1 (valid, user)
  g0[1] = lds;                                              // lds_addr
  g0[2] = (unsigned)g;                                      // global_addr[31:0]
  g0[3] = ((unsigned)(g >> 32) & 0x01FFFFFFu) | (2u << 30); // addr[56:32] | type=2
  tdm_i32x8 g1;
  g1[0] = (int)(2u << 16);                          // wg_mask=0, data_size=4B
  g1[1] = (int)((ndwords & 0xFFFFu) << 16);         // tensor_dim0[15:0]
  g1[2] = (int)(((ndwords >> 16) & 0xFFFFu) | (1u << 16)); // dim0[31:16], tensor_dim1=1
  g1[3] = (int)((ndwords & 0xFFFFu) << 16);         // tile_dim0 = ndwords
  g1[4] = 1;                                        // tile_dim1=1, tile_dim2=0
  g1[5] = (int)ndwords;                             // tensor_dim0_stride lo
  g1[6] = 0;
  g1[7] = 0;
  tdm_i32x4 z4 = {0, 0, 0, 0};
  tdm_i32x8 z8 = {0, 0, 0, 0, 0, 0, 0, 0};
  // 6-arg form (clang-23 / therock headers): (g0, g1, g2, g3, g4, cpol)
  __builtin_amdgcn_tensor_load_to_lds(g0, g1, z4, z4, z8, 0);
}

// ---- pre-swizzle fp32 weights (row-major [fan_out, fan_in]) into bf16 WMMA B tiles ----
__global__ void convert_weights(const float* __restrict__ W, unsigned short* __restrict__ dst,
                                int fanIn, int NT, int total) {
  int t = blockIdx.x * blockDim.x + threadIdx.x;
  if (t >= total) return;
  int i    = t & 15;
  int L    = (t >> 4) & 31;
  int tile = t >> 9;
  int nt   = tile % NT;
  int kt   = tile / NT;
  int gg = L >> 4, r = i >> 1, p = i & 1;
  int K = ((r & 3) << 1) + p + gg * 8 + ((r >> 2) << 4);
  int n = nt * 16 + (L & 15);
  int k = kt * 32 + K;
  dst[t] = f2bf(W[n * fanIn + k]);
}

__global__ void zero_f32(float* __restrict__ a, int n) {
  int t = blockIdx.x * blockDim.x + threadIdx.x;
  if (t < n) a[t] = 0.0f;
}

__global__ void finalize(float* __restrict__ out, const float* __restrict__ x,
                         const float* __restrict__ cnt, int total) {
  int t = blockIdx.x * blockDim.x + threadIdx.x;
  if (t >= total) return;
  float c = cnt[t >> 7];                 // D_IN == 128
  out[t] = (c > 0.0f) ? out[t] / c : x[t];
}

// ---- main edge-MLP kernel: one 16-edge WMMA M-tile per wave ----
__global__ __launch_bounds__(WAVES * 32) void gnn_edges(
    const float* __restrict__ x, const int* __restrict__ eidx,
    const unsigned short* __restrict__ B1w, const unsigned short* __restrict__ B2w,
    const unsigned short* __restrict__ B3w,
    const float* __restrict__ b1, const float* __restrict__ b2, const float* __restrict__ b3,
    float* __restrict__ sums, float* __restrict__ cnt, int E) {

  // per-wave buffer: feats (16x384 bf16, 12KB) then reused for h1 (16x512 bf16, 16KB)
  __shared__ alignas(16) unsigned short bufS[WAVES][16 * H1DIM];
  __shared__ alignas(16) unsigned short h2S[WAVES][16 * H2DIM];
  __shared__ alignas(16) unsigned short B2s[NB2H];   // TDM-staged W2 tiles (64KB)
  __shared__ alignas(16) unsigned short B3s[NB3H];   // TDM-staged W3 tiles (16KB)

  const int lane = threadIdx.x & 31;
  const int wv   = threadIdx.x >> 5;
  const int base = (blockIdx.x * WAVES + wv) * 16;
  const int mrow = lane & 15;
  const int g    = lane >> 4;
  const int col  = lane & 15;

  // ---- kick off TDM DMA of layer-2/3 weights; overlaps gather + layer 1 ----
  if (wv == 0) {
    tdm_copy_1d(B2w, B2s, NB2H / 2);
    tdm_copy_1d(B3w, B3s, NB3H / 2);
  }

  unsigned short* fw = bufS[wv];

  // ---- gather x[obj]|x[pred]|x[sub] -> bf16 LDS tile (16 x 384) ----
  for (int m = 0; m < 16; ++m) {
    int e = base + m;
    #pragma unroll
    for (int s = 0; s < 3; ++s) {
      int node = (e < E) ? eidx[e * 3 + s] : 0;
      float4 v = *(const float4*)(x + (long)node * D_IN + (lane << 2));
      ushort4 h;
      h.x = f2bf(v.x); h.y = f2bf(v.y); h.z = f2bf(v.z); h.w = f2bf(v.w);
      *(ushort4*)(fw + m * FEAT + s * D_IN + (lane << 2)) = h;   // ds_store_b64
    }
  }

  // ---- layer 1: (16x384) @ (384x512), bias + ReLU ----
  // Hoist all A tiles to registers, then h1 may legally overwrite the feats region
  // (LDS ops are in-order within a wave).
  const unsigned short* arow = fw + mrow * FEAT + g * 8;
  v16bf a1[KT1];
  #pragma unroll
  for (int kt = 0; kt < KT1; ++kt) a1[kt] = ld_a(arow + kt * 32);

  unsigned short* h1w = bufS[wv];
  for (int nt = 0; nt < NT1; nt += 2) {     // two accumulators -> independent WMMA chains
    v8f acc0 = {}, acc1 = {};
    #pragma unroll
    for (int kt = 0; kt < KT1; ++kt) {
      v16bf bA = ld_b(B1w + (kt * NT1 + nt) * 512 + lane * 16);
      v16bf bB = ld_b(B1w + (kt * NT1 + nt + 1) * 512 + lane * 16);
      acc0 = wmma_bf16(a1[kt], bA, acc0);
      acc1 = wmma_bf16(a1[kt], bB, acc1);
    }
    float bias0 = b1[nt * 16 + col];
    float bias1 = b1[(nt + 1) * 16 + col];
    #pragma unroll
    for (int j = 0; j < 8; ++j) {
      float r0 = acc0[j] + bias0; r0 = r0 > 0.0f ? r0 : 0.0f;
      float r1 = acc1[j] + bias1; r1 = r1 > 0.0f ? r1 : 0.0f;
      h1w[(j + 8 * g) * H1DIM + nt * 16 + col]       = f2bf(r0);
      h1w[(j + 8 * g) * H1DIM + (nt + 1) * 16 + col] = f2bf(r1);
    }
  }

  // ---- wait for TDM-staged weights, make visible to all waves ----
  if (wv == 0) __builtin_amdgcn_s_wait_tensorcnt(0);
  __syncthreads();

  // ---- layer 2: (16x512) @ (512x64), B from LDS, bias + ReLU ----
  const unsigned short* h1row = h1w + mrow * H1DIM + g * 8;
  v8f acc2[NT2] = {};
  for (int kt = 0; kt < KT2; ++kt) {
    v16bf a = ld_a(h1row + kt * 32);
    #pragma unroll
    for (int nt = 0; nt < NT2; ++nt) {
      v16bf b = ld_b(B2s + (kt * NT2 + nt) * 512 + lane * 16);
      acc2[nt] = wmma_bf16(a, b, acc2[nt]);
    }
  }
  unsigned short* h2w = h2S[wv];
  #pragma unroll
  for (int nt = 0; nt < NT2; ++nt) {
    float bias = b2[nt * 16 + col];
    #pragma unroll
    for (int j = 0; j < 8; ++j) {
      float r = acc2[nt][j] + bias;
      r = r > 0.0f ? r : 0.0f;
      h2w[(j + 8 * g) * H2DIM + nt * 16 + col] = f2bf(r);
    }
  }

  // ---- layer 3: (16x64) @ (64x128), B from LDS, bias, f32 atomic scatter-add ----
  const unsigned short* h2row = h2w + mrow * H2DIM + g * 8;
  v16bf a3[KT3];
  #pragma unroll
  for (int kt = 0; kt < KT3; ++kt) a3[kt] = ld_a(h2row + kt * 32);

  int obj8[8];
  #pragma unroll
  for (int j = 0; j < 8; ++j) {
    int e = base + j + 8 * g;                 // C/D row M = j + 8*(lane/16)
    obj8[j] = (e < E) ? eidx[e * 3] : -1;
  }

  for (int nt = 0; nt < NT3; ++nt) {
    v8f acc = {};
    #pragma unroll
    for (int kt = 0; kt < KT3; ++kt) {
      v16bf b = ld_b(B3s + (kt * NT3 + nt) * 512 + lane * 16);
      acc = wmma_bf16(a3[kt], b, acc);
    }
    float bias = b3[nt * 16 + col];
    #pragma unroll
    for (int j = 0; j < 8; ++j) {
      if (obj8[j] >= 0)
        unsafeAtomicAdd(sums + (long)obj8[j] * D_IN + nt * 16 + col, acc[j] + bias);
    }
  }

  if (lane < 16) {
    int e = base + lane;
    if (e < E) unsafeAtomicAdd(cnt + eidx[e * 3], 1.0f);
  }
}

extern "C" void kernel_launch(void* const* d_in, const int* in_sizes, int n_in,
                              void* d_out, int out_size, void* d_ws, size_t ws_size,
                              hipStream_t stream) {
  const float* x  = (const float*)d_in[0];
  const int* eidx = (const int*)d_in[1];
  const float* W1 = (const float*)d_in[2];
  const float* b1 = (const float*)d_in[3];
  const float* W2 = (const float*)d_in[4];
  const float* b2 = (const float*)d_in[5];
  const float* W3 = (const float*)d_in[6];
  const float* b3 = (const float*)d_in[7];
  float* out = (float*)d_out;

  const int N = in_sizes[0] / D_IN;
  const int E = in_sizes[1] / 3;

  const int nB1 = KT1 * NT1 * 512;   // 196608 halves
  unsigned short* B1w = (unsigned short*)d_ws;
  unsigned short* B2w = B1w + nB1;
  unsigned short* B3w = B2w + NB2H;
  float* cnt = (float*)((char*)d_ws + (size_t)(nB1 + NB2H + NB3H) * 2);

  // accumulation buffers re-zeroed every call (graph-replay deterministic)
  zero_f32<<<(N * D_IN + 255) / 256, 256, 0, stream>>>(out, N * D_IN);
  zero_f32<<<(N + 255) / 256, 256, 0, stream>>>(cnt, N);

  convert_weights<<<(nB1 + 255) / 256, 256, 0, stream>>>(W1, B1w, FEAT,  NT1, nB1);
  convert_weights<<<(NB2H + 255) / 256, 256, 0, stream>>>(W2, B2w, H1DIM, NT2, NB2H);
  convert_weights<<<(NB3H + 255) / 256, 256, 0, stream>>>(W3, B3w, H2DIM, NT3, NB3H);

  int blocks = (E + WAVES * 16 - 1) / (WAVES * 16);
  gnn_edges<<<blocks, WAVES * 32, 0, stream>>>(x, eidx, B1w, B2w, B3w,
                                               b1, b2, b3, out, cnt, E);

  finalize<<<(N * D_IN + 255) / 256, 256, 0, stream>>>(out, x, cnt, N * D_IN);
}
